// Attention_37409165148668
// MI455X (gfx1250) — matrix operand order repaired and checked by reading, not compiled, not run on hardware
//
#include <hip/hip_runtime.h>
#include <stdint.h>

#define Bb 2
#define Ss 2048
#define DIMd 1024
#define Hh 16
#define Dd 64
#define SPv (Ss + 1024)   // padded Vt row stride (512 pad each side)

typedef __bf16 bf16_t;
typedef __attribute__((ext_vector_type(16))) __bf16    v16bf;
typedef __attribute__((ext_vector_type(8)))  float     v8f;
typedef __attribute__((ext_vector_type(8)))  uint32_t  v8u;

static __device__ __forceinline__ uint32_t packbf2(float a, float b) {
  uint32_t ua = __builtin_bit_cast(uint32_t, a);
  uint32_t ub = __builtin_bit_cast(uint32_t, b);
  ua += 0x7FFFu + ((ua >> 16) & 1u);
  ub += 0x7FFFu + ((ub >> 16) & 1u);
  return (ua >> 16) | (ub & 0xFFFF0000u);
}
static __device__ __forceinline__ uint16_t f2bf(float a) {
  uint32_t ua = __builtin_bit_cast(uint32_t, a);
  ua += 0x7FFFu + ((ua >> 16) & 1u);
  return (uint16_t)(ua >> 16);
}
static __device__ __forceinline__ float sigm(float x) { return 1.0f / (1.0f + __expf(-x)); }

// Load a WMMA A-operand (16x32 bf16) half-row: two contiguous b128s.
static __device__ __forceinline__ v16bf ld_ab(const bf16_t* p0, const bf16_t* p1) {
  uint4 a0 = *(const uint4*)p0;
  uint4 a1 = *(const uint4*)p1;
  v8u u = {a0.x, a0.y, a0.z, a0.w, a1.x, a1.y, a1.z, a1.w};
  return __builtin_bit_cast(v16bf, u);
}

// ---------------- f32 -> bf16 convert (pairs) ----------------
__global__ void k_cvt_bf16(const float* __restrict__ src, uint32_t* __restrict__ dst, int n2) {
  int i = blockIdx.x * blockDim.x + threadIdx.x;
  if (i < n2) dst[i] = packbf2(src[2 * i], src[2 * i + 1]);
}

// ---------------- weight swizzle: W[K][N] f32 -> B-operand blobs ----------------
// Blob layout: WB[ntile][kchunk][lane][reg0..7] (u32, each = 2 bf16)
// B-operand (32x16 bf16): lane = n%16 + 16*h ; lane-half h holds K = 16h..16h+15,
// reg r holds K {16h+2r, 16h+2r+1}.
__global__ void k_swizzle_w(const float* __restrict__ W, uint32_t* __restrict__ WB,
                            int K, int N) {
  int i = blockIdx.x * blockDim.x + threadIdx.x;
  int total = (N >> 4) * (K >> 5) * 256;
  if (i >= total) return;
  int r    = i & 7;
  int lane = (i >> 3) & 31;
  int rest = i >> 8;
  int KC   = K >> 5;
  int kc   = rest % KC;
  int nt   = rest / KC;
  int h    = lane >> 4;
  int n    = nt * 16 + (lane & 15);
  int klo  = kc * 32 + 16 * h + 2 * r;
  WB[i] = packbf2(W[(size_t)klo * N + n], W[(size_t)(klo + 1) * N + n]);
}

// ---------------- generic bf16 WMMA GEMM: C[M][N] = A[M][K] * W[K][N] ----------------
// A row-major bf16; W pre-swizzled; each wave computes a 16 x (16*TN) strip.
template <int TN>
__global__ void __launch_bounds__(128) k_gemm(const bf16_t* __restrict__ A,
                                              const uint32_t* __restrict__ WB,
                                              float* __restrict__ C,
                                              int M, int N, int K) {
  int wave = blockIdx.x * (blockDim.x >> 5) + (threadIdx.x >> 5);
  int lane = threadIdx.x & 31;
  int ln = lane & 15, hl = lane >> 4;
  int mtiles = M >> 4;
  int mt = wave % mtiles;
  int ng = wave / mtiles;
  int m0 = mt << 4;
  int nt0 = ng * TN;
  int KC = K >> 5;
  v8f acc[TN];
#pragma unroll
  for (int t = 0; t < TN; ++t) acc[t] = (v8f){};
  const bf16_t* arow = A + (size_t)(m0 + ln) * K;
  for (int kc = 0; kc < KC; ++kc) {
    v16bf av = ld_ab(arow + kc * 32 + 16 * hl, arow + kc * 32 + 16 * hl + 8);
#pragma unroll
    for (int t = 0; t < TN; ++t) {
      const uint32_t* bp = WB + (((size_t)(nt0 + t) * KC + kc) * 32 + lane) * 8;
      uint4 b0 = *(const uint4*)bp;
      uint4 b1 = *(const uint4*)(bp + 4);
      v8u bu = {b0.x, b0.y, b0.z, b0.w, b1.x, b1.y, b1.z, b1.w};
      v16bf bv = __builtin_bit_cast(v16bf, bu);
      acc[t] = __builtin_amdgcn_wmma_f32_16x16x32_bf16(false, av, false, bv,
                                                       (short)0, acc[t], false, false);
    }
  }
#pragma unroll
  for (int t = 0; t < TN; ++t)
#pragma unroll
    for (int r = 0; r < 8; ++r)
      C[(size_t)(m0 + r + 8 * hl) * N + (nt0 + t) * 16 + ln] = acc[t][r];
}

// ---------------- epilogue: scale+RoPE Q (to swizzled B blobs), RoPE K (row major),
//                 value-residual mix V (to transposed padded Vt), sigmoid gate ------
__global__ void k_epilogue(const float* __restrict__ Qf, const float* __restrict__ KVf,
                           const float* __restrict__ Gf, const float* __restrict__ Mixf,
                           const float* __restrict__ vres,
                           uint32_t* __restrict__ QB, uint32_t* __restrict__ Krm,
                           uint16_t* __restrict__ Vt, float* __restrict__ gate) {
  int t = blockIdx.x * blockDim.x + threadIdx.x;   // 2^21 threads
  int i = t & 31;                // d-pair index 0..31
  int s = (t >> 5) & (Ss - 1);
  int h = (t >> 16) & (Hh - 1);
  int b = t >> 20;
  int d = 2 * i;
  size_t row = (size_t)b * Ss + s;

  float ang = (float)s * __powf(10000.0f, -(float)i / 32.0f);
  float c, sn;
  __sincosf(ang, &sn, &c);

  // ---- Q: scale + rope -> swizzled B-operand blob
  const float scale = 0.125f;  // D^-0.5
  float qe = Qf[row * 1024 + h * 64 + d];
  float qo = Qf[row * 1024 + h * 64 + d + 1];
  float qe2 = scale * (qe * c - qo * sn);
  float qo2 = scale * (qo * c + qe * sn);
  int qt = s >> 4;
  int kk = d & 31, chunk = d >> 5;
  int hh = kk >> 4;
  int r = (kk & 15) >> 1;
  int lane = (s & 15) + 16 * hh;
  size_t qboff = ((((size_t)(b * Hh + h) * (Ss / 16) + qt) * 2 + chunk) * 32 + lane) * 8 + r;
  QB[qboff] = packbf2(qe2, qo2);

  // ---- K: rope -> row major [b,h,s,64] bf16
  float ke = KVf[row * 2048 + h * 64 + d];
  float ko = KVf[row * 2048 + h * 64 + d + 1];
  Krm[((size_t)(b * Hh + h) * Ss + s) * 32 + i] = packbf2(ke * c - ko * sn, ko * c + ke * sn);

  // ---- V: residual mix -> transposed padded Vt[b,h,d, 512+s]
  float mixv = sigm(Mixf[row * 16 + h]);
  size_t vr = ((size_t)(b * Hh + h) * Ss + s) * 64 + d;
  float ve = KVf[row * 2048 + 1024 + h * 64 + d];
  float vo = KVf[row * 2048 + 1024 + h * 64 + d + 1];
  ve = ve + mixv * (vres[vr] - ve);
  vo = vo + mixv * (vres[vr + 1] - vo);
  size_t vtb = ((size_t)(b * Hh + h) * 64 + d) * SPv + 512 + s;
  Vt[vtb] = f2bf(ve);
  Vt[vtb + SPv] = f2bf(vo);

  if (i == 0) gate[(size_t)(b * Hh + h) * Ss + s] = sigm(Gf[row * 16 + h]);
}

// ---------------- flash attention: 1 wave per (b,h, 16-query tile) ----------------
__global__ void __launch_bounds__(128) k_attn(const uint32_t* __restrict__ QB,
                                              const bf16_t* __restrict__ Krm,
                                              const bf16_t* __restrict__ Vt,
                                              const float* __restrict__ gate,
                                              const int* __restrict__ ep_ids,
                                              bf16_t* __restrict__ attnO) {
  __shared__ __align__(16) int eps[608];
  int bid = blockIdx.x;
  int nqb = Ss / 64;
  int qb = (bid % nqb) * 64;
  int h = (bid / nqb) % Hh;
  int b = bid / (nqb * Hh);
  for (int idx = threadIdx.x; idx < 608; idx += 128) {
    int j = qb - 512 + idx;
    j = max(0, min(Ss - 1, j));
    eps[idx] = ep_ids[b * Ss + j];
  }
  __syncthreads();

  int wave = threadIdx.x >> 5, lane = threadIdx.x & 31;
  int ln = lane & 15, hl = lane >> 4;
  int qt = (qb >> 4) + wave;
  int qi = qt * 16 + ln;
  int ep_q = eps[qi - qb + 512];

  // Q B-operands (two K=32 chunks over d)
  const uint32_t* qbp = QB + (((size_t)(b * Hh + h) * (Ss / 16) + qt) * 2) * 256;
  v16bf Bq[2];
#pragma unroll
  for (int cc = 0; cc < 2; ++cc) {
    const uint32_t* p = qbp + (size_t)(cc * 32 + lane) * 8;
    uint4 x0 = *(const uint4*)p;
    uint4 x1 = *(const uint4*)(p + 4);
    v8u u = {x0.x, x0.y, x0.z, x0.w, x1.x, x1.y, x1.z, x1.w};
    Bq[cc] = __builtin_bit_cast(v16bf, u);
  }
  const bf16_t* kbase = Krm + (size_t)(b * Hh + h) * Ss * 64;
  const bf16_t* vbase = Vt + (size_t)(b * Hh + h) * 64 * SPv + 512;

  float m = -INFINITY, lsum = 0.f;
  v8f O[4];
#pragma unroll
  for (int t = 0; t < 4; ++t) O[t] = (v8f){};

  for (int p = 0; p < 17; ++p) {
    int j0 = qt * 16 - 512 + p * 32;
    float sv[16];
#pragma unroll
    for (int t = 0; t < 2; ++t) {
      int jt = j0 + 16 * t;
      v8f sacc = (v8f){};
      const bf16_t* krow = kbase + (size_t)(jt + ln) * 64;   // padded region absorbs jt<0
#pragma unroll
      for (int cc = 0; cc < 2; ++cc) {
        v16bf av = ld_ab(krow + cc * 32 + 16 * hl, krow + cc * 32 + 16 * hl + 8);
        sacc = __builtin_amdgcn_wmma_f32_16x16x32_bf16(false, av, false, Bq[cc],
                                                       (short)0, sacc, false, false);
      }
      // branchless mask: vector LDS read of the 8 episode ids this lane needs
      int jb = jt + 8 * hl;                    // key index of reg r=0 (C layout)
      int eidx = jb - qb + 512;                // multiple of 8 -> 32B aligned
      int4 e0 = *(const int4*)&eps[eidx];
      int4 e1 = *(const int4*)&eps[eidx + 4];
      int ev[8] = {e0.x, e0.y, e0.z, e0.w, e1.x, e1.y, e1.z, e1.w};
#pragma unroll
      for (int r = 0; r < 8; ++r) {
        int j = jb + r;
        int ok = (j >= 0) & (j <= qi) & ((qi - j) <= 512) & (ev[r] == ep_q);
        sv[t * 8 + r] = ok ? sacc[r] : -1e30f;
      }
    }
    // online softmax per query column (lane pair l, l+16 share the query)
    float tmax = sv[0];
#pragma unroll
    for (int x = 1; x < 16; ++x) tmax = fmaxf(tmax, sv[x]);
    tmax = fmaxf(tmax, __shfl_xor(tmax, 16, 32));
    float mn = fmaxf(m, tmax);
    float alpha = __expf(m - mn);
    float pv[16];
    float psum = 0.f;
#pragma unroll
    for (int x = 0; x < 16; ++x) { pv[x] = __expf(sv[x] - mn); psum += pv[x]; }
    psum += __shfl_xor(psum, 16, 32);
    lsum = lsum * alpha + psum;
    m = mn;
#pragma unroll
    for (int t = 0; t < 4; ++t)
#pragma unroll
      for (int r = 0; r < 8; ++r) O[t][r] *= alpha;

    // Build PV B-operand: lane-half h needs ALL keys 16h..16h+15 -> one xor-16 exchange.
    float lo[8], hi[8];
#pragma unroll
    for (int r = 0; r < 8; ++r) {
      float x = hl ? pv[r] : pv[8 + r];
      float y = __shfl_xor(x, 16, 32);
      lo[r] = hl ? y : pv[r];
      hi[r] = hl ? pv[8 + r] : y;
    }
    v8u bpu;
#pragma unroll
    for (int r = 0; r < 4; ++r) {
      bpu[r]     = packbf2(lo[2 * r], lo[2 * r + 1]);
      bpu[4 + r] = packbf2(hi[2 * r], hi[2 * r + 1]);
    }
    v16bf Bp = __builtin_bit_cast(v16bf, bpu);

    // O^T[d-tile] += V^T(16d x 32k) * P(32k x 16q)
#pragma unroll
    for (int t = 0; t < 4; ++t) {
      const bf16_t* vrow = vbase + (size_t)(t * 16 + ln) * SPv + j0;
      v16bf av = ld_ab(vrow + 16 * hl, vrow + 16 * hl + 8);
      O[t] = __builtin_amdgcn_wmma_f32_16x16x32_bf16(false, av, false, Bp,
                                                     (short)0, O[t], false, false);
    }
  }

  float f = gate[(size_t)(b * Hh + h) * Ss + qi] / lsum;
#pragma unroll
  for (int t = 0; t < 4; ++t) {
    uint4 ov;
    ov.x = packbf2(O[t][0] * f, O[t][1] * f);
    ov.y = packbf2(O[t][2] * f, O[t][3] * f);
    ov.z = packbf2(O[t][4] * f, O[t][5] * f);
    ov.w = packbf2(O[t][6] * f, O[t][7] * f);
    size_t o = ((size_t)b * Ss + qi) * 1024 + h * 64 + t * 16 + 8 * hl;
    *(uint4*)(attnO + o) = ov;
  }
}

// ------------------------------- host side -------------------------------
extern "C" void kernel_launch(void* const* d_in, const int* in_sizes, int n_in,
                              void* d_out, int out_size, void* d_ws, size_t ws_size,
                              hipStream_t stream) {
  (void)in_sizes; (void)n_in; (void)out_size; (void)ws_size;
  const float* tokens = (const float*)d_in[0];
  const float* vres   = (const float*)d_in[1];
  const int*   epi    = (const int*)d_in[2];
  const float* Wq     = (const float*)d_in[3];
  const float* Wkv    = (const float*)d_in[4];
  const float* Wo     = (const float*)d_in[5];
  const float* Wg     = (const float*)d_in[6];
  const float* Wmix   = (const float*)d_in[7];

  char* ws = (char*)d_ws;
  size_t off = 0;
  auto alloc = [&](size_t bytes) { size_t o = off; off += (bytes + 255) & ~(size_t)255; return o; };

  size_t tokbf = alloc((size_t)Bb * Ss * DIMd * 2);
  size_t wqb   = alloc((size_t)1024 * 1024 * 2);
  size_t wkvb  = alloc((size_t)1024 * 2048 * 2);
  size_t wob   = alloc((size_t)1024 * 1024 * 2);
  size_t wgb   = alloc((size_t)1024 * 16 * 2);
  size_t wmb   = alloc((size_t)1024 * 16 * 2);
  size_t qbb   = alloc((size_t)Bb * Hh * (Ss / 16) * 2 * 32 * 8 * 4);
  (void)alloc(65536);                                   // pad before K (absorbs j<0 reads)
  size_t krm   = alloc((size_t)Bb * Hh * Ss * 64 * 2 + 8192);
  size_t vt    = alloc((size_t)Bb * Hh * 64 * SPv * 2);
  size_t gat   = alloc((size_t)Bb * Hh * Ss * 4);
  size_t gf    = alloc((size_t)Bb * Ss * Hh * 4);
  size_t mixf  = alloc((size_t)Bb * Ss * Hh * 4);
  size_t qf    = alloc((size_t)Bb * Ss * 1024 * 4);     // attnO (bf16, 8MB) aliases this later
  size_t kvf   = alloc((size_t)Bb * Ss * 2048 * 4);

  bf16_t*   tokb = (bf16_t*)(ws + tokbf);
  uint32_t* QBp  = (uint32_t*)(ws + qbb);
  uint32_t* KrmU = (uint32_t*)(ws + krm);
  bf16_t*   Kbf  = (bf16_t*)(ws + krm);
  uint16_t* VtU  = (uint16_t*)(ws + vt);
  bf16_t*   Vtb  = (bf16_t*)(ws + vt);
  float*    gatp = (float*)(ws + gat);
  float*    Qf   = (float*)(ws + qf);
  float*    KVf  = (float*)(ws + kvf);
  float*    Gf   = (float*)(ws + gf);
  float*    Mixf = (float*)(ws + mixf);
  bf16_t*   attO = (bf16_t*)(ws + qf);                  // alias: Qf dead after epilogue

  // 1) convert tokens to bf16
  int n2 = Bb * Ss * DIMd / 2;
  k_cvt_bf16<<<(n2 + 255) / 256, 256, 0, stream>>>(tokens, (uint32_t*)tokb, n2);

  // 2) swizzle weights into B-operand blobs
  auto swz = [&](const float* W, size_t o, int K, int N) {
    int total = (N >> 4) * (K >> 5) * 256;
    k_swizzle_w<<<(total + 255) / 256, 256, 0, stream>>>(W, (uint32_t*)(ws + o), K, N);
  };
  swz(Wq,   wqb,  1024, 1024);
  swz(Wkv,  wkvb, 1024, 2048);
  swz(Wo,   wob,  1024, 1024);
  swz(Wg,   wgb,  1024, 16);
  swz(Wmix, wmb,  1024, 16);

  // 3) projection GEMMs (bf16 WMMA, f32 out)
  k_gemm<4><<<1024, 128, 0, stream>>>(tokb, (uint32_t*)(ws + wqb),  Qf,   4096, 1024, 1024);
  k_gemm<4><<<2048, 128, 0, stream>>>(tokb, (uint32_t*)(ws + wkvb), KVf,  4096, 2048, 1024);
  k_gemm<1><<<64,   128, 0, stream>>>(tokb, (uint32_t*)(ws + wgb),  Gf,   4096, 16,   1024);
  k_gemm<1><<<64,   128, 0, stream>>>(tokb, (uint32_t*)(ws + wmb),  Mixf, 4096, 16,   1024);

  // 4) RoPE / scale / value-mix / gate epilogue
  k_epilogue<<<8192, 256, 0, stream>>>(Qf, KVf, Gf, Mixf, vres, QBp, KrmU, VtU, gatp);

  // 5) flash attention (WMMA S^T and O^T)
  k_attn<<<Bb * Hh * (Ss / 64), 128, 0, stream>>>(QBp, Kbf, Vtb, gatp, epi, attO);

  // 6) output projection into d_out (f32)
  k_gemm<4><<<1024, 128, 0, stream>>>(attO, (uint32_t*)(ws + wob), (float*)d_out,
                                      4096, 1024, 1024);
}
